// GNN_63668595196006
// MI455X (gfx1250) — matrix operand order repaired and checked by reading, not compiled, not run on hardware
//
#include <hip/hip_runtime.h>
#include <math.h>

// ---------------------------------------------------------------------------
// MI455X / gfx1250 implementation.
//   - All large GEMMs (conv MLPs, policy/value heads) via v_wmma_f32_16x16x32_f16:
//     fp32 weights streamed from HBM once (~700MB -> ~30us @ 23.3TB/s),
//     converted to f16 in-register/LDS, fp32 accumulate.
//   - Aligned-only WMMA kernel: M%128==0, N%32==0, K%32==0 -> branch-free K loop.
//   - W tile staged via LDS (f16) and shared by all 8 waves of the block.
//   - Irregular shapes (K=1 rank-1, N=1 dot+tanh) use dedicated kernels.
// ---------------------------------------------------------------------------

typedef _Float16 v16h __attribute__((ext_vector_type(16)));
typedef _Float16 v4h  __attribute__((ext_vector_type(4)));
typedef float    v8f  __attribute__((ext_vector_type(8)));

union U16H { v16h v; _Float16 h[16]; };
union U4H  { v4h  v; _Float16 h[4];  };
union U8F  { v8f  v; float    f[8];  };

#define NORM_EPS 1e-5f

// ---------------------------------------------------------------------------
__global__ void fill_kernel(float* __restrict__ p, float val, int count) {
  int i = blockIdx.x * blockDim.x + threadIdx.x;
  if (i < count) p[i] = val;
}

// H[b,dst,c] = x[b,dst,c] + sum_src relu(x[b,src,c] + obs[b,src*n+dst]*We[c] + be[c])
__global__ void gine_aggr_kernel(const float* __restrict__ obs,
                                 const float* __restrict__ X,
                                 const float* __restrict__ We,
                                 const float* __restrict__ be,
                                 float* __restrict__ H,
                                 int B, int n, int C) {
  int idx = blockIdx.x * blockDim.x + threadIdx.x;
  int total = B * n * C;
  if (idx >= total) return;
  int c   = idx % C;
  int dst = (idx / C) % n;
  int b   = idx / (C * n);
  const float* xb = X + (size_t)b * n * C;
  const float* ob = obs + (size_t)b * n * n;
  float we = We[c], beb = be[c];
  float acc = xb[dst * C + c];
  for (int src = 0; src < n; ++src) {
    float m = xb[src * C + c] + ob[src * n + dst] * we + beb;
    acc += m > 0.f ? m : 0.f;
  }
  H[idx] = acc;
}

// Rank-1 "GEMM" for conv0 layer0 (K == 1): Y[row, c] = h[row] * W[c] + b[c].
__global__ void rank1_kernel(const float* __restrict__ h,
                             const float* __restrict__ W,
                             const float* __restrict__ bias,
                             float* __restrict__ Y, int R, int C) {
  int i = blockIdx.x * blockDim.x + threadIdx.x;
  if (i >= R * C) return;
  int c = i % C;
  Y[i] = h[i / C] * W[c] + bias[c];
}

// Value head output (N == 1): out[row] = tanh(dot(A[row,:], W[0,:]) + b[0]).
__global__ void dot_tanh_kernel(const float* __restrict__ A,
                                const float* __restrict__ W,
                                const float* __restrict__ bias,
                                float* __restrict__ out, int K) {
  int row = blockIdx.x;
  const float* ar = A + (size_t)row * K;
  float s = 0.f;
  for (int k = threadIdx.x; k < K; k += blockDim.x) s += ar[k] * W[k];
  __shared__ float sh[256];
  sh[threadIdx.x] = s;
  __syncthreads();
  for (int ofs = 128; ofs > 0; ofs >>= 1) {
    if ((int)threadIdx.x < ofs) sh[threadIdx.x] += sh[threadIdx.x + ofs];
    __syncthreads();
  }
  if (threadIdx.x == 0) out[row] = tanhf(sh[0] + bias[0]);
}

// ---------------------------------------------------------------------------
// Aligned WMMA GEMM: Y[M,N] = A[M,K] @ W[N,K]^T + bias.
// Requires M%128==0, N%32==0, K%32==0. grid = (N/32, M/128), block = 256.
// Each wave owns a 16(M) x 32(N) strip = two 16x16 WMMA accumulators.
// W tile (32 cols x 32 K) staged cooperatively into LDS as f16 per K-step.
// ---------------------------------------------------------------------------
__global__ void __launch_bounds__(256)
gemm_f16_wmma_kernel(const float* __restrict__ A,
                     const float* __restrict__ W,
                     const float* __restrict__ bias,
                     float* __restrict__ Y,
                     int M, int N, int K, int act) {
  __shared__ alignas(32) _Float16 Bs[32 * 32];   // [col 0..31][k 0..31]

  const int tid  = threadIdx.x;
  const int lane = tid & 31;
  const int wave = tid >> 5;
  const int g    = lane >> 4;       // half-wave group
  const int r    = lane & 15;
  const int nblk = blockIdx.x * 32;             // 32 output cols per block
  const int mrow0 = blockIdx.y * 128 + wave * 16;

  // cooperative B staging: 256 threads x 4 floats = 32x32 fp32 tile
  const int scol = tid >> 3;                    // 0..31
  const int sk   = (tid & 7) * 4;               // 0,4,...,28
  const float* wsrc  = W + (size_t)(nblk + scol) * K + sk;
  const float* arowp = A + (size_t)(mrow0 + r) * K;

  v8f acc0 = {}, acc1 = {};
  for (int kb = 0; kb < K; kb += 32) {
    // issue global W load before the barrier: overlaps previous stage compute
    float4 wv = *reinterpret_cast<const float4*>(wsrc + kb);
    __syncthreads();                            // previous stage fully consumed
    U4H wh;
    wh.h[0] = (_Float16)wv.x; wh.h[1] = (_Float16)wv.y;
    wh.h[2] = (_Float16)wv.z; wh.h[3] = (_Float16)wv.w;
    *reinterpret_cast<v4h*>(&Bs[scol * 32 + sk]) = wh.v;
    __syncthreads();                            // stage ready

    // A 16x32 f16 fragment: lane holds row (mrow0+r); K runs
    // [kb+g*16, +16) contiguous -> two float4 pairs, branch-free.
    U16H a;
    {
      const float4* p0 = reinterpret_cast<const float4*>(arowp + kb + g * 16);
      float4 a0 = p0[0], a1 = p0[1];
      const float4* p1 = reinterpret_cast<const float4*>(arowp + kb + g * 16 + 8);
      float4 a2 = p1[0], a3 = p1[1];
      a.h[0]  = (_Float16)a0.x; a.h[1]  = (_Float16)a0.y;
      a.h[2]  = (_Float16)a0.z; a.h[3]  = (_Float16)a0.w;
      a.h[4]  = (_Float16)a1.x; a.h[5]  = (_Float16)a1.y;
      a.h[6]  = (_Float16)a1.z; a.h[7]  = (_Float16)a1.w;
      a.h[8]  = (_Float16)a2.x; a.h[9]  = (_Float16)a2.y;
      a.h[10] = (_Float16)a2.z; a.h[11] = (_Float16)a2.w;
      a.h[12] = (_Float16)a3.x; a.h[13] = (_Float16)a3.y;
      a.h[14] = (_Float16)a3.z; a.h[15] = (_Float16)a3.w;
    }
    // B 32x16 f16 fragments from LDS: lane holds column, 16 contiguous K
    // starting at g*16 (32B-aligned).
    v16h b0 = *reinterpret_cast<const v16h*>(&Bs[r * 32 + g * 16]);
    v16h b1 = *reinterpret_cast<const v16h*>(&Bs[(16 + r) * 32 + g * 16]);

    // two independent WMMAs share the A fragment (fills hazard slots)
    acc0 = __builtin_amdgcn_wmma_f32_16x16x32_f16(false, a.v, false, b0,
                                                  (short)0, acc0, false, false);
    acc1 = __builtin_amdgcn_wmma_f32_16x16x32_f16(false, a.v, false, b1,
                                                  (short)0, acc1, false, false);
  }

  // D layout: VGPR r2, lane -> (M = r2 + 8*g, N = r)
  U8F c0, c1; c0.v = acc0; c1.v = acc1;
  const int col0 = nblk + r, col1 = nblk + 16 + r;
  const float bv0 = bias[col0], bv1 = bias[col1];
#pragma unroll
  for (int r2 = 0; r2 < 8; ++r2) {
    int rowm = mrow0 + r2 + 8 * g;
    float v0 = c0.f[r2] + bv0;
    float v1 = c1.f[r2] + bv1;
    if (act == 1) { v0 = tanhf(v0); v1 = tanhf(v1); }
    Y[(size_t)rowm * N + col0] = v0;
    Y[(size_t)rowm * N + col1] = v1;
  }
}

// Per-feature mean/var over R rows (training-mode BN). grid = C, block = 256.
__global__ void bn_stats_kernel(const float* __restrict__ Y, int R, int C,
                                float* __restrict__ mean, float* __restrict__ var) {
  int c = blockIdx.x;
  float s = 0.f, s2 = 0.f;
  for (int row = threadIdx.x; row < R; row += blockDim.x) {
    float v = Y[(size_t)row * C + c];
    s += v; s2 += v * v;
  }
  __shared__ float sh[256], sh2[256];
  sh[threadIdx.x] = s; sh2[threadIdx.x] = s2;
  __syncthreads();
  for (int ofs = 128; ofs > 0; ofs >>= 1) {
    if ((int)threadIdx.x < ofs) {
      sh[threadIdx.x]  += sh[threadIdx.x + ofs];
      sh2[threadIdx.x] += sh2[threadIdx.x + ofs];
    }
    __syncthreads();
  }
  if (threadIdx.x == 0) {
    float m = sh[0] / (float)R;
    mean[c] = m;
    var[c]  = sh2[0] / (float)R - m * m;
  }
}

__global__ void bn_apply_relu_kernel(float* __restrict__ Y, int R, int C,
                                     const float* __restrict__ mean,
                                     const float* __restrict__ var,
                                     const float* __restrict__ gw,
                                     const float* __restrict__ nb) {
  int i = blockIdx.x * blockDim.x + threadIdx.x;
  if (i >= R * C) return;
  int c = i % C;
  float rs = rsqrtf(var[c] + NORM_EPS);
  float v = (Y[i] - mean[c]) * rs * gw[c] + nb[c];
  Y[i] = v > 0.f ? v : 0.f;
}

// Fused LayerNorm(+affine)+ReLU over feature dim, in place. grid = rows.
__global__ void ln_relu_kernel(float* __restrict__ Y, int C,
                               const float* __restrict__ gw,
                               const float* __restrict__ nb) {
  int row = blockIdx.x;
  float* yr = Y + (size_t)row * C;
  float s = 0.f, s2 = 0.f;
  for (int c = threadIdx.x; c < C; c += blockDim.x) {
    float v = yr[c]; s += v; s2 += v * v;
  }
  __shared__ float sh[256], sh2[256];
  __shared__ float sm, srs;
  sh[threadIdx.x] = s; sh2[threadIdx.x] = s2;
  __syncthreads();
  for (int ofs = 128; ofs > 0; ofs >>= 1) {
    if ((int)threadIdx.x < ofs) {
      sh[threadIdx.x]  += sh[threadIdx.x + ofs];
      sh2[threadIdx.x] += sh2[threadIdx.x + ofs];
    }
    __syncthreads();
  }
  if (threadIdx.x == 0) {
    float m = sh[0] / (float)C;
    sm  = m;
    srs = rsqrtf(sh2[0] / (float)C - m * m + NORM_EPS);
  }
  __syncthreads();
  for (int c = threadIdx.x; c < C; c += blockDim.x) {
    float v = (yr[c] - sm) * srs * gw[c] + nb[c];
    yr[c] = v > 0.f ? v : 0.f;
  }
}

// JumpingKnowledge('max') running accumulator.
__global__ void max_into_kernel(float* __restrict__ JK,
                                const float* __restrict__ X, int count) {
  int i = blockIdx.x * blockDim.x + threadIdx.x;
  if (i < count) {
    float a = JK[i], b = X[i];
    JK[i] = a > b ? a : b;
  }
}

// ---------------------------------------------------------------------------
extern "C" void kernel_launch(void* const* d_in, const int* in_sizes, int n_in,
                              void* d_out, int out_size, void* d_ws, size_t ws_size,
                              hipStream_t stream) {
  (void)in_sizes; (void)n_in; (void)out_size; (void)ws_size;
  const int B = 128, n = 64, HID = 32;
  const int Rn = B * n;              // 8192 node-rows
  const int DH = n * HID;            // 2048
  const int PH = 8192;               // hidden width of policy/value MLPs
  const int NPOL = n * (n - 1) / 2;  // 2016

  const float* obs = (const float*)d_in[0];
  float* pol_out = (float*)d_out;               // [128, 2016]
  float* val_out = pol_out + (size_t)B * NPOL;  // [128, 1]

  // workspace layout (floats)
  float* ws   = (float*)d_ws;
  float* X    = ws;                 // [8192, 32] node features (C=1 for conv0)
  float* H    = X  + (size_t)Rn * HID;
  float* Yb   = H  + (size_t)Rn * HID;
  float* JK   = Yb + (size_t)Rn * HID;   // [128, 2048]
  float* mean = JK + (size_t)Rn * HID;
  float* var  = mean + PH;
  float* P1   = var  + PH;          // [128, 8192]
  float* P2   = P1 + (size_t)B * PH;

  auto gemm = [&](const float* A, int iW, int ib, float* Y,
                  int M, int N, int K, int act) {
    dim3 grid(N / 32, M / 128);     // aligned-only kernel
    gemm_f16_wmma_kernel<<<grid, 256, 0, stream>>>(
        A, (const float*)d_in[iW], (const float*)d_in[ib], Y, M, N, K, act);
  };
  auto bn = [&](float* Y, int ig, int inb) {
    bn_stats_kernel<<<HID, 256, 0, stream>>>(Y, Rn, HID, mean, var);
    bn_apply_relu_kernel<<<(Rn * HID + 255) / 256, 256, 0, stream>>>(
        Y, Rn, HID, mean, var, (const float*)d_in[ig], (const float*)d_in[inb]);
  };

  // init: x = ones [B,n,1]; JK = -inf
  fill_kernel<<<(Rn + 255) / 256, 256, 0, stream>>>(X, 1.0f, Rn);
  fill_kernel<<<(Rn * HID + 255) / 256, 256, 0, stream>>>(JK, -1e30f, Rn * HID);

  // ---- 4x GINEConv (flatten order per conv: We, be, [W,b,g,nb]x3, [W,b]) ----
  int C = 1;
  for (int l = 0; l < 4; ++l) {
    const int base = 1 + 16 * l;
    int tot = B * n * C;
    gine_aggr_kernel<<<(tot + 255) / 256, 256, 0, stream>>>(
        obs, X, (const float*)d_in[base], (const float*)d_in[base + 1], H, B, n, C);
    if (l == 0) {  // K == 1: rank-1 update instead of a degenerate GEMM
      rank1_kernel<<<(Rn * HID + 255) / 256, 256, 0, stream>>>(
          H, (const float*)d_in[base + 2], (const float*)d_in[base + 3],
          Yb, Rn, HID);
    } else {
      gemm(H, base + 2, base + 3, Yb, Rn, HID, HID, 0);
    }
    bn(Yb, base + 4, base + 5);
    gemm(Yb, base + 6,  base + 7,  X,  Rn, HID, HID, 0); bn(X,  base + 8,  base + 9);
    gemm(X,  base + 10, base + 11, Yb, Rn, HID, HID, 0); bn(Yb, base + 12, base + 13);
    gemm(Yb, base + 14, base + 15, X,  Rn, HID, HID, 0);            // last: plain
    max_into_kernel<<<(Rn * HID + 255) / 256, 256, 0, stream>>>(JK, X, Rn * HID);
    C = HID;
  }

  // ---- policy head: [2048 -> 8192 -> 8192 -> 2016], LN+ReLU on hidden ----
  const int pb = 1 + 64;   // 65
  gemm(JK, pb + 0, pb + 1, P1, B, PH, DH, 0);
  ln_relu_kernel<<<B, 256, 0, stream>>>(P1, PH, (const float*)d_in[pb + 2],
                                        (const float*)d_in[pb + 3]);
  gemm(P1, pb + 4, pb + 5, P2, B, PH, PH, 0);
  ln_relu_kernel<<<B, 256, 0, stream>>>(P2, PH, (const float*)d_in[pb + 6],
                                        (const float*)d_in[pb + 7]);
  gemm(P2, pb + 8, pb + 9, pol_out, B, NPOL, PH, 0);   // 2016 = 63*32

  // ---- value head: [2048 -> 8192 -> 8192 -> 1], tanh output ----
  const int vb = pb + 10;  // 75
  gemm(JK, vb + 0, vb + 1, P1, B, PH, DH, 0);
  ln_relu_kernel<<<B, 256, 0, stream>>>(P1, PH, (const float*)d_in[vb + 2],
                                        (const float*)d_in[vb + 3]);
  gemm(P1, vb + 4, vb + 5, P2, B, PH, PH, 0);
  ln_relu_kernel<<<B, 256, 0, stream>>>(P2, PH, (const float*)d_in[vb + 6],
                                        (const float*)d_in[vb + 7]);
  dot_tanh_kernel<<<B, 256, 0, stream>>>(P2, (const float*)d_in[vb + 8],
                                         (const float*)d_in[vb + 9], val_out, PH);
}
